// E42LinearTiedCell_10806137717001
// MI455X (gfx1250) — matrix operand order repaired and checked by reading, not compiled, not run on hardware
//
#include <hip/hip_runtime.h>
#include <hip/hip_bf16.h>
#include <math.h>

// ---------------------------------------------------------------------------
// Problem constants (match reference)
// ---------------------------------------------------------------------------
#define TT 2048
#define BB 16
#define DD 1024
#define SPECTRAL_RADIUS 0.999f
#define EPSF 1e-8f
#define N_POWER_ITERS 3

#define NT_TILES (DD / 16)       // 64 N-tiles of 16 columns
#define KC_CHUNKS (DD / 32)      // 32 K-chunks of 32
#define RECUR_BLOCKS NT_TILES    // 64 single-wave workgroups
#define HB_ELEMS (BB * DD)       // 16384 elements per h ping-pong slot

#define GEMM_NT_PER_WAVE 8       // each wave computes 16 x 128 of Wx

// Workspace layout (bytes)
#define WS_SCALE_OFF 0u
#define WS_BPACK_OFF 256u                                     // __bf16[DD*DD] = 2 MB
#define WS_HB_OFF    (WS_BPACK_OFF + 2u * DD * DD)            // __bf16[2*BB*DD] = 64 KB
#define WS_CTR_OFF   (WS_HB_OFF + 2u * 2u * BB * DD)          // unsigned[2048]

typedef __attribute__((ext_vector_type(16))) __bf16 v16bf;
typedef __attribute__((ext_vector_type(8)))  __bf16 v8bf;
typedef __attribute__((ext_vector_type(8)))  float  v8f;

__device__ __forceinline__ v8f wmma_bf16(v16bf a, v16bf b, v8f c) {
  // (neg_a, A, neg_b, B, c_mod, C, reuse_a, reuse_b)
  return __builtin_amdgcn_wmma_f32_16x16x32_bf16(false, a, false, b, (short)0, c,
                                                 false, false);
}

// ---------------------------------------------------------------------------
// Kernel 0: spectral normalization scale via power iteration (1 block, 1024 thr)
// ---------------------------------------------------------------------------
__device__ __forceinline__ float block_reduce_sum(float x, float* red) {
  const int t = threadIdx.x;
  __syncthreads();              // protect 'red' re-use
  red[t] = x;
  __syncthreads();
  for (int s = 512; s > 0; s >>= 1) {
    if (t < s) red[t] += red[t + s];
    __syncthreads();
  }
  return red[0];
}

__global__ __launch_bounds__(1024)
void e42_spectral_kernel(const float* __restrict__ W, const float* __restrict__ u,
                         float* __restrict__ scale_out) {
  __shared__ float su[DD];
  __shared__ float sv[DD];
  __shared__ float sw[DD];
  __shared__ float red[DD];
  const int t = threadIdx.x;

  float uv = u[t];
  float n0 = sqrtf(block_reduce_sum(uv * uv, red));
  su[t] = uv / (n0 + EPSF);
  __syncthreads();

  for (int it = 0; it < N_POWER_ITERS; ++it) {
    // v = W^T @ u  (coalesced: thread t reads column t)
    float av = 0.f;
    for (int i = 0; i < DD; ++i) av += W[(size_t)i * DD + t] * su[i];
    float nv = sqrtf(block_reduce_sum(av * av, red));
    sv[t] = av / (nv + EPSF);
    __syncthreads();
    // u = W @ v  (thread t reads row t)
    float au = 0.f;
    for (int j = 0; j < DD; ++j) au += W[(size_t)t * DD + j] * sv[j];
    sw[t] = au;  // keep last raw W@v for sigma
    float nu = sqrtf(block_reduce_sum(au * au, red));
    su[t] = au / (nu + EPSF);
    __syncthreads();
  }
  // sigma = |u_f . (W @ v_f)|
  float sig = fabsf(block_reduce_sum(su[t] * sw[t], red));
  if (t == 0) *scale_out = SPECTRAL_RADIUS / (sig + EPSF);
}

// ---------------------------------------------------------------------------
// Kernel 1: pack Wn = W*scale into bf16 WMMA B-operand layout.
// Bpack[nt][kc][lane][e] = Wn[n][k],  n = nt*16 + lane%16,
//   k = kc*32 + (e<8 ? (lane/16)*8 + e : 16 + (lane/16)*8 + (e-8))
// ---------------------------------------------------------------------------
__global__ __launch_bounds__(256)
void e42_pack_kernel(const float* __restrict__ W, const float* __restrict__ scale,
                     __bf16* __restrict__ Bpack) {
  const int idx = blockIdx.x * 256 + threadIdx.x;  // DD*DD threads
  const int n = idx >> 10;
  const int k = idx & (DD - 1);
  const float w = W[(size_t)n * DD + k] * (*scale);
  const int nt = n >> 4, ln = n & 15;
  const int kc = k >> 5, kk = k & 31;
  const int lh = (kk >> 3) & 1, e = (kk & 7) + ((kk >> 4) << 3);
  const int lane = lh * 16 + ln;
  Bpack[(((size_t)(nt * KC_CHUNKS + kc) * 32) + lane) * 16 + e] = (__bf16)w;
}

// ---------------------------------------------------------------------------
// Kernel 2: init h[0] = h0 (fp32 out), hb ping slot 0 (bf16), zero barriers.
// ---------------------------------------------------------------------------
__global__ __launch_bounds__(256)
void e42_init_kernel(const float* __restrict__ h0, float* __restrict__ hreg,
                     __bf16* __restrict__ hb, unsigned* __restrict__ ctr) {
  const int idx = blockIdx.x * 256 + threadIdx.x;  // 16384 threads
  if (idx < HB_ELEMS) {
    const float v = h0[idx];
    hreg[idx] = v;              // h[0] slot of output h-tensor
    hb[idx] = (__bf16)v;        // bf16 state, ping slot 0
  }
  if (idx < TT) ctr[idx] = 0u;  // step barriers
}

// ---------------------------------------------------------------------------
// Kernel 3: Wx = x @ Wn^T + b, stored into h-region slots 1..T.
// Block = 256 threads = 8 waves; wave w handles m-tile (blockIdx.x*8+w),
// 8 consecutive N-tiles per wave (blockIdx.y selects group of 128 columns).
// A (x row fragment, fp32->bf16) is loaded once per k-chunk and reused for
// all 8 WMMAs -> 8 independent accumulator chains, x traffic cut to 8x.
// ---------------------------------------------------------------------------
__global__ __launch_bounds__(256)
void e42_gemm_kernel(const float* __restrict__ x, const float* __restrict__ bias,
                     const __bf16* __restrict__ Bpack, float* __restrict__ hreg) {
  const int lane = threadIdx.x & 31;
  const int wave = threadIdx.x >> 5;
  const int lm = lane & 15, lh = lane >> 4;
  const int mtile = blockIdx.x * 8 + wave;           // 0..2047 (rows of X / 16)
  const int ntb = blockIdx.y * GEMM_NT_PER_WAVE;     // first of 8 N-tiles

  const float* xrow = x + (size_t)(mtile * 16 + lm) * DD;

  v8f acc[GEMM_NT_PER_WAVE];
#pragma unroll
  for (int j = 0; j < GEMM_NT_PER_WAVE; ++j) acc[j] = (v8f)(0.f);

  for (int kc = 0; kc < KC_CHUNKS; ++kc) {
    const int k0 = kc * 32 + lh * 8;
    // A lane layout: e<8 -> K=k0+e ; e>=8 -> K=k0+16+(e-8)
    const float4 fa = *(const float4*)(xrow + k0);
    const float4 fb = *(const float4*)(xrow + k0 + 4);
    const float4 fc = *(const float4*)(xrow + k0 + 16);
    const float4 fd = *(const float4*)(xrow + k0 + 20);
    if (kc + 1 < KC_CHUNKS) __builtin_prefetch(xrow + k0 + 32, 0, 1);
    v16bf a;
    a[0] = (__bf16)fa.x;  a[1] = (__bf16)fa.y;  a[2] = (__bf16)fa.z;  a[3] = (__bf16)fa.w;
    a[4] = (__bf16)fb.x;  a[5] = (__bf16)fb.y;  a[6] = (__bf16)fb.z;  a[7] = (__bf16)fb.w;
    a[8] = (__bf16)fc.x;  a[9] = (__bf16)fc.y;  a[10] = (__bf16)fc.z; a[11] = (__bf16)fc.w;
    a[12] = (__bf16)fd.x; a[13] = (__bf16)fd.y; a[14] = (__bf16)fd.z; a[15] = (__bf16)fd.w;
#pragma unroll
    for (int j = 0; j < GEMM_NT_PER_WAVE; ++j) {
      const v16bf bm = *(const v16bf*)(Bpack +
          ((size_t)((ntb + j) * KC_CHUNKS + kc) * 32 + lane) * 16);
      acc[j] = wmma_bf16(a, bm, acc[j]);
    }
  }

#pragma unroll
  for (int j = 0; j < GEMM_NT_PER_WAVE; ++j) {
    const int n = (ntb + j) * 16 + lm;
    const float bv = bias[n];
#pragma unroll
    for (int r = 0; r < 8; ++r) {
      // C layout: lanes 0-15 -> M=r ; lanes 16-31 -> M=8+r
      const int grow = mtile * 16 + lh * 8 + r;            // global row t*B+batch
      hreg[(size_t)(grow + BB) * DD + n] = acc[j][r] + bv; // slot h[t+1]
    }
  }
}

// ---------------------------------------------------------------------------
// Kernel 4: sequential recurrence. 64 blocks x 1 wave. Block nt owns 16 cols.
// Per step: h_t = Wx_t + h_{t-1} @ Wn^T ; out_t = h_t^2 * sigmoid(h_t);
// device-wide atomic barrier between steps.
// ---------------------------------------------------------------------------
__global__ __launch_bounds__(32)
void e42_recur_kernel(const __bf16* __restrict__ Bpack, __bf16* __restrict__ hb,
                      float* __restrict__ hreg, float* __restrict__ out,
                      unsigned* __restrict__ ctr) {
  const int nt = blockIdx.x;
  const int lane = threadIdx.x;
  const int lm = lane & 15, lh = lane >> 4;

  // Stage this block's 1024x16 Wn^T slice in LDS (32 KB), WMMA-ready.
  __shared__ __attribute__((aligned(32))) __bf16 lB[KC_CHUNKS * 32 * 16];
  for (int kc = 0; kc < KC_CHUNKS; ++kc) {
    const v16bf tmp = *(const v16bf*)(Bpack +
        ((size_t)(nt * KC_CHUNKS + kc) * 32 + lane) * 16);
    *(v16bf*)(&lB[(kc * 32 + lane) * 16]) = tmp;
  }
  __syncthreads();

  const int n = nt * 16 + lm;

  for (int t = 0; t < TT; ++t) {
    const __bf16* hcur = hb + (size_t)(t & 1) * HB_ELEMS;
    __bf16* hnxt = hb + (size_t)((t + 1) & 1) * HB_ELEMS;
    float* hout = hreg + (size_t)(t + 1) * HB_ELEMS;   // Wx_t lives here, h_t will too

    v8f acc0, acc1;
#pragma unroll
    for (int r = 0; r < 8; ++r) {
      acc0[r] = hout[(size_t)(lh * 8 + r) * DD + n];   // C init = Wx_t tile
      acc1[r] = 0.f;
    }

#pragma unroll 2
    for (int kc = 0; kc < KC_CHUNKS; ++kc) {
      const int k0 = kc * 32 + lh * 8;
      union { v16bf v; v8bf h[2]; } A;
      A.h[0] = *(const v8bf*)(hcur + (size_t)lm * DD + k0);
      A.h[1] = *(const v8bf*)(hcur + (size_t)lm * DD + k0 + 16);
      union { v16bf v; v8bf h[2]; } Bm;
      Bm.h[0] = *(const v8bf*)(&lB[(kc * 32 + lane) * 16]);
      Bm.h[1] = *(const v8bf*)(&lB[(kc * 32 + lane) * 16 + 8]);
      if (kc & 1) acc1 = wmma_bf16(A.v, Bm.v, acc1);   // 2-deep chain for ILP
      else        acc0 = wmma_bf16(A.v, Bm.v, acc0);
    }

    float* orow = out + (size_t)t * HB_ELEMS;
#pragma unroll
    for (int r = 0; r < 8; ++r) {
      const int m = lh * 8 + r;
      const float h = acc0[r] + acc1[r];
      hout[(size_t)m * DD + n] = h;                    // overwrite Wx with h_t
      const float sg = 1.f / (1.f + __expf(-h));
      orow[(size_t)m * DD + n] = h * h * sg;           // h * silu(h)
      hnxt[(size_t)m * DD + n] = (__bf16)h;            // bf16 state for t+1
    }

    if (t != TT - 1) {
      __threadfence();
      __syncthreads();
      if (lane == 0) {
        atomicAdd(&ctr[t], 1u);
        while (__hip_atomic_load(&ctr[t], __ATOMIC_RELAXED,
                                 __HIP_MEMORY_SCOPE_AGENT) < (unsigned)RECUR_BLOCKS) {
          __builtin_amdgcn_s_sleep(1);
        }
      }
      __syncthreads();
      __threadfence();
    }
  }
}

// ---------------------------------------------------------------------------
// Host launcher
// ---------------------------------------------------------------------------
extern "C" void kernel_launch(void* const* d_in, const int* in_sizes, int n_in,
                              void* d_out, int out_size, void* d_ws, size_t ws_size,
                              hipStream_t stream) {
  (void)in_sizes; (void)n_in; (void)out_size; (void)ws_size;
  const float* x  = (const float*)d_in[0];  // [T,B,D]
  const float* h0 = (const float*)d_in[1];  // [B,D]
  const float* W  = (const float*)d_in[2];  // [D,D]
  const float* b  = (const float*)d_in[3];  // [D]
  const float* u  = (const float*)d_in[4];  // [D]

  float* out  = (float*)d_out;                         // [T,B,D]
  float* hreg = out + (size_t)TT * BB * DD;            // [T+1,B,D]

  char* ws = (char*)d_ws;
  float*    scale = (float*)(ws + WS_SCALE_OFF);
  __bf16*   Bpack = (__bf16*)(ws + WS_BPACK_OFF);
  __bf16*   hb    = (__bf16*)(ws + WS_HB_OFF);
  unsigned* ctr   = (unsigned*)(ws + WS_CTR_OFF);

  e42_spectral_kernel<<<1, 1024, 0, stream>>>(W, u, scale);
  e42_pack_kernel<<<(DD * DD) / 256, 256, 0, stream>>>(W, scale, Bpack);
  e42_init_kernel<<<HB_ELEMS / 256, 256, 0, stream>>>(h0, hreg, hb, ctr);
  e42_gemm_kernel<<<dim3((TT * BB / 16) / 8, DD / (16 * GEMM_NT_PER_WAVE)), 256, 0,
                    stream>>>(x, b, Bpack, hreg);
  e42_recur_kernel<<<RECUR_BLOCKS, 32, 0, stream>>>(Bpack, hb, hreg, out, ctr);
}
